// SurvivalModule_70325794505241
// MI455X (gfx1250) — matrix operand order repaired and checked by reading, not compiled, not run on hardware
//
#include <hip/hip_runtime.h>

// ---------------- problem constants (reference: D=1024, T=512, B=16384) ----
constexpr int kD  = 1024;   // embedding dim (K of GEMM)
constexpr int kT  = 512;    // n_future (N of GEMM)
constexpr int kMT = 32;     // rows of x per workgroup (M tile)
constexpr int kKC = 32;     // K chunk = WMMA K for bf16
constexpr int kNI = kD / kKC;  // 32 K-iterations

typedef __attribute__((ext_vector_type(16))) __bf16 v16bf;
typedef __attribute__((ext_vector_type(8)))  float  v8f;

union BF16Frag { v16bf v; unsigned int u[8]; };

__device__ __forceinline__ unsigned short f2bf(float f) {
    unsigned int u = __float_as_uint(f);
    unsigned int r = u + 0x7FFFu + ((u >> 16) & 1u);   // round-to-nearest-even
    return (unsigned short)(r >> 16);
}

// ---- CDNA5 async global->LDS copy (ASYNCcnt-tracked), GV addressing mode ----
__device__ __forceinline__ void async_copy_b128(unsigned lds_addr,
                                                unsigned long long gaddr) {
    asm volatile("global_load_async_to_lds_b128 %0, %1, off"
                 :: "v"(lds_addr), "v"(gaddr) : "memory");
}
__device__ __forceinline__ void wait_async_le8() {
    asm volatile("s_wait_asynccnt 0x8" ::: "memory");
}
__device__ __forceinline__ void wait_async_0() {
    asm volatile("s_wait_asynccnt 0x0" ::: "memory");
}

// ---------------- kernel 1: Wh fp32 -> bf16 into workspace ----------------
__global__ void wh_to_bf16_kernel(const float* __restrict__ w,
                                  unsigned short* __restrict__ o, int n) {
    int i = (blockIdx.x * blockDim.x + threadIdx.x) * 8;
    if (i + 7 >= n) { for (int j = 0; j < 8 && i + j < n; ++j) o[i+j] = f2bf(w[i+j]); return; }
    #pragma unroll
    for (int j = 0; j < 8; ++j) o[i + j] = f2bf(w[i + j]);
}

// ---------------- kernel 2: fused GEMM(bf16 WMMA) + bias/ReLU + scan ------
__global__ __launch_bounds__(256)
void survival_main_kernel(const float* __restrict__ x,
                          const unsigned short* __restrict__ whb,  // [T][D] bf16
                          const float* __restrict__ bh,
                          const float* __restrict__ Wb,
                          const float* __restrict__ bb,
                          float* __restrict__ out) {
    __shared__ unsigned short lsB[2][kT * kKC];   // Wh tiles, 2 x 32 KB (double buffer)
    __shared__ unsigned short lsA[2][kMT * kKC];  // x  tiles, 2 x 2 KB (bf16)
    __shared__ float wsum[kMT][8];                // per-row per-wave segment totals
    __shared__ float bsum[kMT][8];                // base-term partials
    __shared__ float baseL[kMT];                  // base[row] = x.Wb + bb

    const int tid  = threadIdx.x;
    const int wave = tid >> 5;        // 0..7 -> owns cols [wave*64, wave*64+64)
    const int lane = tid & 31;
    const int lh   = lane & 15;       // lane within 16-lane half
    const int hi   = lane >> 4;       // 0 or 1 (which half of the wave)
    const long b0  = (long)blockIdx.x * kMT;

    // ---- base term: x[b].Wb + bb, pure fp32 (also warms L2 with our x rows) ----
    {
        int r = tid >> 3, s = tid & 7;
        const float* xr = x  + (b0 + r) * kD + s * 128;
        const float* wr = Wb + s * 128;
        float a = 0.f;
        #pragma unroll 4
        for (int i = 0; i < 128; ++i) a += xr[i] * wr[i];
        bsum[r][s] = a;
    }
    __syncthreads();
    if (tid < kMT) {
        float a = bb[0];
        #pragma unroll
        for (int j = 0; j < 8; ++j) a += bsum[tid][j];
        baseL[tid] = a;
    }

    // stage helpers ---------------------------------------------------------
    // B: Wh_bf16[t][k1..k1+31] for all 512 t, 8 async b128 per thread
    auto stage_B = [&](int k1, int buf) {
        const int t0 = tid * 2;
        unsigned long long g0 =
            (unsigned long long)(uintptr_t)(whb + (size_t)t0 * kD + k1);
        unsigned long long g1 = g0 + (unsigned long long)kD * 2;
        unsigned l0 = (unsigned)(uintptr_t)&lsB[buf][t0 * kKC];
        unsigned l1 = l0 + kKC * 2;
        #pragma unroll
        for (int q = 0; q < 4; ++q) {
            async_copy_b128(l0 + q * 16, g0 + q * 16);
            async_copy_b128(l1 + q * 16, g1 + q * 16);
        }
    };
    // A: x[b0..b0+31][k1..k1+31] fp32 -> bf16 (4 elems/thread, sync path)
    auto stage_A = [&](int k1, int buf) {
        int m = tid >> 3, kq = (tid & 7) * 4;
        const float4 f = *(const float4*)(x + (b0 + m) * kD + k1 + kq);
        unsigned int lo  = (unsigned int)f2bf(f.x) | ((unsigned int)f2bf(f.y) << 16);
        unsigned int hi2 = (unsigned int)f2bf(f.z) | ((unsigned int)f2bf(f.w) << 16);
        uint2 pk; pk.x = lo; pk.y = hi2;
        *(uint2*)&lsA[buf][m * kKC + kq] = pk;
    };

    // ---- accumulators: 2 M-subtiles x 4 N-subtiles, f32 16x16 C tiles ----
    v8f acc[2][4];
    #pragma unroll
    for (int ms = 0; ms < 2; ++ms)
        #pragma unroll
        for (int ns = 0; ns < 4; ++ns)
            #pragma unroll
            for (int e = 0; e < 8; ++e) acc[ms][ns][e] = 0.f;

    // ---- software-pipelined K loop (double-buffered LDS) ----
    stage_B(0, 0);
    stage_A(0, 0);
    for (int it = 0; it < kNI; ++it) {
        const int p = it & 1;
        if (it + 1 < kNI) {
            stage_B((it + 1) * kKC, 1 - p);   // async: overlaps this chunk's WMMAs
            stage_A((it + 1) * kKC, 1 - p);
            wait_async_le8();                 // previous stage (buf p) fully landed
        } else {
            wait_async_0();
        }
        __syncthreads();                      // buf p visible to all waves

        // A fragments: this code's contiguous k order (B matches); dword d holds k pair
        // k_base(d) = 2d + (hi ? 16 : 0)
        BF16Frag afr[2];
        #pragma unroll
        for (int ms = 0; ms < 2; ++ms) {
            int row = ms * 16 + lh;
            const unsigned int* pa = (const unsigned int*)&lsA[p][row * kKC];
            #pragma unroll
            for (int dd = 0; dd < 8; ++dd) {
                int kb = 2 * dd + (hi ? 16 : 0);
                afr[ms].u[dd] = pa[kb >> 1];
            }
        }
        // B fragments + WMMA: same contiguous k order as A; k_base(d) = 2d + (hi?16:0)
        #pragma unroll
        for (int ns = 0; ns < 4; ++ns) {
            int n = wave * 64 + ns * 16 + lh;
            const unsigned int* pb = (const unsigned int*)&lsB[p][n * kKC];
            BF16Frag bfr;
            #pragma unroll
            for (int dd = 0; dd < 8; ++dd) {
                int kb = 2 * dd + (hi ? 16 : 0);
                bfr.u[dd] = pb[kb >> 1];
            }
            acc[0][ns] = __builtin_amdgcn_wmma_f32_16x16x32_bf16(
                false, afr[0].v, false, bfr.v, (short)0, acc[0][ns], false, false);
            acc[1][ns] = __builtin_amdgcn_wmma_f32_16x16x32_bf16(
                false, afr[1].v, false, bfr.v, (short)0, acc[1][ns], false, false);
        }
        __syncthreads();                      // done reading buf p before overwrite
    }

    // ---- epilogue: bias + ReLU + inclusive scan along T ----
    // C layout: vgpr e, lanes 0-15 -> row ms*16+e, lanes 16-31 -> row ms*16+e+8.
    #pragma unroll
    for (int ms = 0; ms < 2; ++ms) {
        float carry[8];
        #pragma unroll
        for (int e = 0; e < 8; ++e) carry[e] = 0.f;

        #pragma unroll
        for (int ns = 0; ns < 4; ++ns) {
            int n = wave * 64 + ns * 16 + lh;
            float bv = bh[n];
            #pragma unroll
            for (int e = 0; e < 8; ++e) {
                float v = acc[ms][ns][e] + bv;
                v = v > 0.f ? v : 0.f;
                #pragma unroll
                for (int off = 1; off < 16; off <<= 1) {   // 16-lane inclusive scan
                    float t = __shfl_up(v, off, 16);
                    if (lh >= off) v += t;
                }
                v += carry[e];
                carry[e] = __shfl(v, 15, 16);              // broadcast segment total
                acc[ms][ns][e] = v;
            }
        }
        if (lh == 0) {
            #pragma unroll
            for (int e = 0; e < 8; ++e) {
                int row = ms * 16 + e + hi * 8;
                wsum[row][wave] = carry[e];
            }
        }
    }
    __syncthreads();

    // ---- cross-wave prefix offset + base term, nontemporal streaming store ----
    #pragma unroll
    for (int ms = 0; ms < 2; ++ms) {
        #pragma unroll
        for (int e = 0; e < 8; ++e) {
            int row = ms * 16 + e + hi * 8;
            float off = baseL[row];
            for (int ww = 0; ww < wave; ++ww) off += wsum[row][ww];
            #pragma unroll
            for (int ns = 0; ns < 4; ++ns) {
                int n = wave * 64 + ns * 16 + lh;
                __builtin_nontemporal_store(acc[ms][ns][e] + off,
                                            out + (b0 + row) * kT + n);
            }
        }
    }
}

// ---------------------------------------------------------------------------
extern "C" void kernel_launch(void* const* d_in, const int* in_sizes, int n_in,
                              void* d_out, int out_size, void* d_ws, size_t ws_size,
                              hipStream_t stream) {
    const float* x  = (const float*)d_in[0];   // [B, D]
    const float* Wh = (const float*)d_in[1];   // [T, D]
    const float* bh = (const float*)d_in[2];   // [T]
    const float* Wb = (const float*)d_in[3];   // [1, D]
    const float* bb = (const float*)d_in[4];   // [1]
    float* out = (float*)d_out;
    unsigned short* whb = (unsigned short*)d_ws;  // bf16 Wh, 1 MB

    const int nW = kT * kD;  // 524288
    wh_to_bf16_kernel<<<dim3(nW / (256 * 8)), dim3(256), 0, stream>>>(Wh, whb, nW);

    const int B = in_sizes[0] / kD;  // 16384
    survival_main_kernel<<<dim3(B / kMT), dim3(256), 0, stream>>>(
        x, whb, bh, Wb, bb, out);
}
